// CentroidsSimilarity_56702158242288
// MI455X (gfx1250) — compile-verified, matrix operand order checked by reading, not run on hardware
//
#include <hip/hip_runtime.h>
#include <hip/hip_bf16.h>

// ---------------------------------------------------------------------------
// ProtoNet output: softmax over negative squared distances to class centroids.
// N=16384 rows, C=128 classes, D=1024 dims.
// logits[n,c] = 2*dot(e_n, cent_c) - ||cent_c||^2   (||e_n||^2 cancels in softmax)
// ---------------------------------------------------------------------------

#define N_DIM 16384
#define C_DIM 128
#define D_DIM 1024

typedef __attribute__((ext_vector_type(16))) __bf16   v16bf;
typedef __attribute__((ext_vector_type(4)))  __bf16   v4bf;
typedef __attribute__((ext_vector_type(8)))  float    v8f;
typedef __attribute__((ext_vector_type(4)))  float    f32x4;
typedef __attribute__((ext_vector_type(4)))  unsigned u32x4;

// workspace layout (bytes)
#define CS_OFF   0u                       // centroid sums  [C][D] f32  (512 KB)
#define CNT_OFF  (C_DIM * D_DIM * 4u)     // counts         [C]    f32  (512 B)
#define SQC_OFF  (CNT_OFF + 512u)         // ||cent||^2     [C]    f32  (512 B)
#define CB_OFF   (SQC_OFF + 512u)         // centroids bf16 [C][D]      (256 KB)

__device__ __forceinline__ unsigned short f2bf_rne(float f) {
  unsigned u = __builtin_bit_cast(unsigned, f);
  unsigned r = u + 0x7FFFu + ((u >> 16) & 1u);
  return (unsigned short)(r >> 16);
}

// ---------------------------------------------------------------------------
// Kernel 1: segment-sum accumulation. One wave (32 lanes) per row.
// ---------------------------------------------------------------------------
__global__ void centroid_accum(const float* __restrict__ emb,
                               const float* __restrict__ y_true,
                               float* __restrict__ cent_sum,
                               float* __restrict__ counts) {
  const int wave = threadIdx.x >> 5;
  const int lane = threadIdx.x & 31;
  const int row  = blockIdx.x * 8 + wave;
  if (row >= N_DIM) return;

  // recover label from one-hot row: each lane checks 4 columns
  const float* yrow = y_true + (size_t)row * C_DIM;
  int lab = -1;
#pragma unroll
  for (int j = 0; j < 4; ++j) {
    int c = lane + 32 * j;
    if (yrow[c] > 0.5f) lab = c;
  }
#pragma unroll
  for (int off = 16; off >= 1; off >>= 1)
    lab = max(lab, __shfl_xor(lab, off, 32));
  if (lab < 0) return;  // defensive (one-hot guarantees a 1)

  if (lane == 0) atomicAdd(counts + lab, 1.0f);

  const float* arow = emb + (size_t)row * D_DIM;
  float* crow = cent_sum + (size_t)lab * D_DIM;
#pragma unroll 4
  for (int d = lane; d < D_DIM; d += 32)
    atomicAdd(crow + d, arow[d]);
}

// ---------------------------------------------------------------------------
// Kernel 2: finalize centroids (divide_no_nan), emit bf16 copy + ||c||^2.
// One block (256 threads) per class.
// ---------------------------------------------------------------------------
__global__ void centroid_finalize(const float* __restrict__ cent_sum,
                                  const float* __restrict__ counts,
                                  unsigned short* __restrict__ cent_bf,
                                  float* __restrict__ sq_c) {
  __shared__ float red[256];
  const int c   = blockIdx.x;
  const int tid = threadIdx.x;
  const float cnt = counts[c];
  const float inv = (cnt > 0.5f) ? (1.0f / cnt) : 0.0f;

  const float* src = cent_sum + (size_t)c * D_DIM;
  unsigned short* dst = cent_bf + (size_t)c * D_DIM;

  float sq = 0.0f;
#pragma unroll
  for (int d = tid; d < D_DIM; d += 256) {
    float v = src[d] * inv;
    dst[d] = f2bf_rne(v);
    sq += v * v;
  }
  red[tid] = sq;
  __syncthreads();
#pragma unroll
  for (int s = 128; s >= 1; s >>= 1) {
    if (tid < s) red[tid] += red[tid + s];
    __syncthreads();
  }
  if (tid == 0) sq_c[c] = red[0];
}

// ---------------------------------------------------------------------------
// Kernel 3: WMMA cross-term + fused softmax.
// Block = 256 threads = 8 waves; block covers a 16-row x 128-col output tile,
// wave w owns cols [16w, 16w+16). K-loop over D in steps of 32 (bf16 WMMA),
// unrolled x2 so two independent load/convert/WMMA streams interleave.
//
// 16-bit A-fragment layout (ISA 7.12.2): lane L holds row M=L%16,
//   K = (L/16)*8 + {0..7} in regs 0..3 and +16 in regs 4..7.
// B mirrors this with N=L%16 (B read column-major == centroid rows).
// C/D layout: lane L, reg i -> M = i + 8*(L/16), N = L%16.
// ---------------------------------------------------------------------------
__global__ void gemm_softmax(const float* __restrict__ emb,
                             const unsigned short* __restrict__ cent_bf,
                             const float* __restrict__ sq_c,
                             float* __restrict__ out) {
  __shared__ float lds_logits[16 * C_DIM];  // 8 KB

  const int wave     = threadIdx.x >> 5;
  const int lane     = threadIdx.x & 31;
  const int row_base = blockIdx.x * 16;
  const int col_base = wave * 16;
  const int mn       = lane & 15;        // M (for A) / N (for B,C)
  const int kb       = (lane >> 4) * 8;  // K sub-offset by lane half

  const float*          arow = emb     + (size_t)(row_base + mn) * D_DIM;
  const unsigned short* brow = cent_bf + (size_t)(col_base + mn) * D_DIM;

  v8f acc = {};
  union AU { v16bf v; v4bf w[4]; };
  union BU { v16bf v; u32x4 q[2]; };

#pragma unroll 2
  for (int k0 = 0; k0 < D_DIM; k0 += 32) {
    // prefetch next A chunk (speculative; dropped if past end of row)
    __builtin_prefetch(arow + k0 + 32, 0, 0);

    AU a;
    f32x4 f0 = *(const f32x4*)(arow + k0 + kb);
    f32x4 f1 = *(const f32x4*)(arow + k0 + kb + 4);
    f32x4 f2 = *(const f32x4*)(arow + k0 + kb + 16);
    f32x4 f3 = *(const f32x4*)(arow + k0 + kb + 20);
    // fptrunc f32->bf16 (RNE): backend picks native packed cvt if available
    a.w[0] = __builtin_convertvector(f0, v4bf);
    a.w[1] = __builtin_convertvector(f1, v4bf);
    a.w[2] = __builtin_convertvector(f2, v4bf);
    a.w[3] = __builtin_convertvector(f3, v4bf);

    BU b;
    b.q[0] = *(const u32x4*)(brow + k0 + kb);
    b.q[1] = *(const u32x4*)(brow + k0 + kb + 16);

    acc = __builtin_amdgcn_wmma_f32_16x16x32_bf16(
        /*neg_a=*/false, a.v, /*neg_b=*/false, b.v,
        /*c_mod=*/(short)0, acc, /*reuse_a=*/false, /*reuse_b=*/false);
  }

  // logits = 2*cross - ||c||^2  -> LDS tile
  const float sq    = sq_c[col_base + mn];
  const int   rhalf = (lane >> 4) * 8;
#pragma unroll
  for (int i = 0; i < 8; ++i)
    lds_logits[(rhalf + i) * C_DIM + col_base + mn] = 2.0f * acc[i] - sq;
  __syncthreads();

  // softmax: 16 lanes per row, 8 cols per lane, half-wave shuffle reductions
  const int r = threadIdx.x >> 4;  // 0..15
  const int g = threadIdx.x & 15;
  float vals[8];
  float m = -1e30f;
#pragma unroll
  for (int j = 0; j < 8; ++j) {
    vals[j] = lds_logits[r * C_DIM + g + 16 * j];
    m = fmaxf(m, vals[j]);
  }
#pragma unroll
  for (int off = 8; off >= 1; off >>= 1)
    m = fmaxf(m, __shfl_xor(m, off, 16));

  float s = 0.0f;
#pragma unroll
  for (int j = 0; j < 8; ++j) {
    vals[j] = __expf(vals[j] - m);
    s += vals[j];
  }
#pragma unroll
  for (int off = 8; off >= 1; off >>= 1)
    s += __shfl_xor(s, off, 16);

  const float invs = 1.0f / s;
  float* orow = out + (size_t)(row_base + r) * C_DIM;
#pragma unroll
  for (int j = 0; j < 8; ++j)
    __builtin_nontemporal_store(vals[j] * invs, orow + g + 16 * j);
}

// ---------------------------------------------------------------------------
extern "C" void kernel_launch(void* const* d_in, const int* in_sizes, int n_in,
                              void* d_out, int out_size, void* d_ws, size_t ws_size,
                              hipStream_t stream) {
  const float* emb    = (const float*)d_in[0];  // [N, D] f32
  const float* y_true = (const float*)d_in[1];  // [N, C] f32 one-hot
  float* out = (float*)d_out;                   // [N, C] f32

  char* ws = (char*)d_ws;
  float*          cent_sum = (float*)(ws + CS_OFF);
  float*          counts   = (float*)(ws + CNT_OFF);
  float*          sqc      = (float*)(ws + SQC_OFF);
  unsigned short* cent_bf  = (unsigned short*)(ws + CB_OFF);

  // zero centroid sums + counts (graph-capturable stream op)
  hipMemsetAsync(ws, 0, CNT_OFF + 512u, stream);

  centroid_accum<<<N_DIM / 8, 256, 0, stream>>>(emb, y_true, cent_sum, counts);
  centroid_finalize<<<C_DIM, 256, 0, stream>>>(cent_sum, counts, cent_bf, sqc);
  gemm_softmax<<<N_DIM / 16, 256, 0, stream>>>(emb, cent_bf, sqc, out);
}
